// Glm4MoeExpertLayersTorch_1657857376741
// MI455X (gfx1250) — matrix operand order repaired and checked by reading, not compiled, not run on hardware
//
#include <hip/hip_runtime.h>
#include <hip/hip_bf16.h>

typedef __bf16 bf16;
typedef __attribute__((ext_vector_type(16))) __bf16 v16bf;
typedef __attribute__((ext_vector_type(8)))  __bf16 v8bf;
typedef __attribute__((ext_vector_type(4)))  __bf16 v4bf;
typedef __attribute__((ext_vector_type(8)))  float   v8f;
typedef __attribute__((ext_vector_type(4)))  float   v4f;
typedef __attribute__((ext_vector_type(4)))  int     v4i;

// address-space qualified element types for the async global->LDS builtin
typedef __attribute__((address_space(1))) v4i ga_v4i;   // global
typedef __attribute__((address_space(3))) v4i ls_v4i;   // LDS

#define NEXP   8
#define TWOI   3072
#define HDIM   2048
#define IDIM   1536
#define TDIM   16384

// GEMM tiling
#define BM 128
#define BN 64
#define BK 64
#define KP (BK + 8)   // padded LDS K-stride in elements (144 B, 16B-aligned rows)

// async-copy instructions issued per thread per staged tile
#define CPT1 8        // GEMM1: A(4) + G(2) + U(2)
#define CPT2 6        // GEMM2: A(4) + B(2)

// ---------------------------------------------------------------------------
// gfx1250 async global->LDS path (guarded; falls back to load+ds_store)
// ---------------------------------------------------------------------------
#if defined(__has_builtin) && __has_builtin(__builtin_amdgcn_global_load_async_to_lds_b128)
#define HAS_ASYNC_LDS 1
#else
#define HAS_ASYNC_LDS 0
#endif

__device__ __forceinline__ void copy_b128(const bf16* __restrict__ g, bf16* l)
{
#if HAS_ASYNC_LDS
    __builtin_amdgcn_global_load_async_to_lds_b128(
        (ga_v4i*)g, (ls_v4i*)l, 0, 0);
#else
    *(v8bf*)l = *(const v8bf*)g;
#endif
}

#if HAS_ASYNC_LDS
#if defined(__has_builtin) && __has_builtin(__builtin_amdgcn_s_wait_asynccnt)
#define WAIT_ASYNC(n) __builtin_amdgcn_s_wait_asynccnt(n)
#else
#define WAIT_ASYNC(n) asm volatile("s_wait_asynccnt %0" ::"i"(n))
#endif
#else
#define WAIT_ASYNC(n) ((void)0)
#endif

// ---------------------------------------------------------------------------
// fp32 -> bf16 conversion (one-time; optional per-expert slice offset)
// ---------------------------------------------------------------------------
__global__ __launch_bounds__(256)
void cvt_f32_to_bf16(const float* __restrict__ src, bf16* __restrict__ dst,
                     const int* __restrict__ eidx, long long slice, long long n)
{
    long long base = eidx ? (long long)(*eidx) * slice : 0ll;
    long long i = ((long long)blockIdx.x * blockDim.x + threadIdx.x) * 4;
    if (i + 3 < n) {
        v4f x = *(const v4f*)(src + base + i);
        v4bf o;
        o[0] = (bf16)x[0]; o[1] = (bf16)x[1];
        o[2] = (bf16)x[2]; o[3] = (bf16)x[3];
        *(v4bf*)(dst + i) = o;
    }
}

// ---------------------------------------------------------------------------
// Fragment builders (LDS -> VGPR per CDNA5 WMMA layouts, wave32)
// A 16x32 bf16: lane L -> row = L&15; hs = L>>4; K = {hs*8..hs*8+7, hs*8+16..hs*8+23}
// B 32x16 bf16: lane L -> col = L&15; hs = L>>4; K = {hs*8..hs*8+7, hs*8+16..hs*8+23}
// ---------------------------------------------------------------------------
__device__ __forceinline__ v16bf frag_a(const bf16* __restrict__ s, int row, int ks, int hs)
{
    const bf16* p = s + row * KP + ks + hs * 8;
    v8bf lo = *(const v8bf*)(p);
    v8bf hi = *(const v8bf*)(p + 16);
    return __builtin_shufflevector(lo, hi, 0,1,2,3,4,5,6,7,8,9,10,11,12,13,14,15);
}

__device__ __forceinline__ v16bf frag_b(const bf16* __restrict__ s, int col, int ks, int hs)
{
    const bf16* p = s + col * KP + ks + hs * 8;
    v8bf lo = *(const v8bf*)(p);
    v8bf hi = *(const v8bf*)(p + 16);
    return __builtin_shufflevector(lo, hi, 0,1,2,3,4,5,6,7,8,9,10,11,12,13,14,15);
}

#define WMMA_BF16(A, B, C) \
    __builtin_amdgcn_wmma_f32_16x16x32_bf16(false, (A), false, (B), (short)0, (C), false, false)

// ---------------------------------------------------------------------------
// Tile staging (async global->LDS when available)
// ---------------------------------------------------------------------------
__device__ __forceinline__ void stage1(const bf16* __restrict__ X,
                                       const bf16* __restrict__ Wgu,
                                       bf16* sA, bf16* sG, bf16* sU,
                                       int m0, int n0, int kk, int tid)
{
#pragma unroll
    for (int c = tid; c < BM * BK / 8; c += 256) {     // 4 per thread
        int r = c >> 3, ck = (c & 7) * 8;
        copy_b128(X + (long long)(m0 + r) * HDIM + kk + ck, sA + r * KP + ck);
    }
#pragma unroll
    for (int c = tid; c < BN * BK / 8; c += 256) {     // 2+2 per thread
        int r = c >> 3, ck = (c & 7) * 8;
        copy_b128(Wgu + (long long)(n0 + r) * HDIM + kk + ck,        sG + r * KP + ck);
        copy_b128(Wgu + (long long)(IDIM + n0 + r) * HDIM + kk + ck, sU + r * KP + ck);
    }
}

__device__ __forceinline__ void stage2(const bf16* __restrict__ A,
                                       const bf16* __restrict__ W,
                                       bf16* sA, bf16* sB,
                                       int m0, int n0, int kk, int tid)
{
#pragma unroll
    for (int c = tid; c < BM * BK / 8; c += 256) {
        int r = c >> 3, ck = (c & 7) * 8;
        copy_b128(A + (long long)(m0 + r) * IDIM + kk + ck, sA + r * KP + ck);
    }
#pragma unroll
    for (int c = tid; c < BN * BK / 8; c += 256) {
        int r = c >> 3, ck = (c & 7) * 8;
        copy_b128(W + (long long)(n0 + r) * IDIM + kk + ck, sB + r * KP + ck);
    }
}

// ---------------------------------------------------------------------------
// GEMM1 + fused SwiGLU:
//   gate = X @ Wgu[n,:]^T , up = X @ Wgu[I+n,:]^T , Hid = silu(gate)*up  (bf16)
// Double-buffered LDS (async pipeline). Dynamic LDS: 2*(BM+2*BN)*KP*2 = 73728 B
// ---------------------------------------------------------------------------
__global__ __launch_bounds__(256)
void gemm1_swiglu(const bf16* __restrict__ X,    // [T, H]
                  const bf16* __restrict__ Wgu,  // [2I, H]
                  bf16* __restrict__ Hid)        // [T, I]
{
    extern __shared__ char smem[];
    bf16* sA = (bf16*)smem;              // [2][BM*KP]
    bf16* sG = sA + 2 * BM * KP;         // [2][BN*KP]
    bf16* sU = sG + 2 * BN * KP;         // [2][BN*KP]

    const int tid  = threadIdx.x;
    const int lane = tid & 31;
    const int wave = tid >> 5;      // 0..7
    const int wm   = wave >> 1;     // 0..3 : 32-row band
    const int wn   = wave & 1;      // 0..1 : 32-col band
    const int l15  = lane & 15;
    const int hs   = lane >> 4;

    const int m0 = blockIdx.y * BM;
    const int n0 = blockIdx.x * BN;

    v8f accG[2][2], accU[2][2];
#pragma unroll
    for (int a = 0; a < 2; ++a)
#pragma unroll
        for (int b = 0; b < 2; ++b) { accG[a][b] = (v8f)0.0f; accU[a][b] = (v8f)0.0f; }

    const int NT = HDIM / BK;    // 32
    stage1(X, Wgu, sA, sG, sU, m0, n0, 0, tid);

    for (int it = 0; it < NT; ++it) {
        const int cur = it & 1;
        bf16* cA = sA + cur * BM * KP;
        bf16* cG = sG + cur * BN * KP;
        bf16* cU = sU + cur * BN * KP;

        if (it + 1 < NT) {
            stage1(X, Wgu, sA + (cur ^ 1) * BM * KP, sG + (cur ^ 1) * BN * KP,
                   sU + (cur ^ 1) * BN * KP, m0, n0, (it + 1) * BK, tid);
            WAIT_ASYNC(CPT1);            // tile `it` complete (in-order)
        } else {
            WAIT_ASYNC(0);
        }
        __syncthreads();

#pragma unroll
        for (int ks = 0; ks < BK; ks += 32) {
            v16bf af[2];
#pragma unroll
            for (int mt = 0; mt < 2; ++mt)
                af[mt] = frag_a(cA, wm * 32 + mt * 16 + l15, ks, hs);
#pragma unroll
            for (int nt = 0; nt < 2; ++nt) {
                v16bf bg = frag_b(cG, wn * 32 + nt * 16 + l15, ks, hs);
                v16bf bu = frag_b(cU, wn * 32 + nt * 16 + l15, ks, hs);
#pragma unroll
                for (int mt = 0; mt < 2; ++mt) {
                    accG[mt][nt] = WMMA_BF16(af[mt], bg, accG[mt][nt]);
                    accU[mt][nt] = WMMA_BF16(af[mt], bu, accU[mt][nt]);
                }
            }
        }
        __syncthreads();    // buffer `cur` may be refilled next iteration
    }

    // SwiGLU epilogue + bf16 store.
    // C/D layout: lane L, vgpr e -> (M = (L>>4)*8 + e, N = L&15)
#pragma unroll
    for (int mt = 0; mt < 2; ++mt)
#pragma unroll
        for (int nt = 0; nt < 2; ++nt)
#pragma unroll
            for (int e = 0; e < 8; ++e) {
                float g = accG[mt][nt][e];
                float u = accU[mt][nt][e];
                float h = (g / (1.0f + __expf(-g))) * u;   // silu(g) * u
                int row = m0 + wm * 32 + mt * 16 + hs * 8 + e;
                int col = n0 + wn * 32 + nt * 16 + l15;
                Hid[(long long)row * IDIM + col] = (bf16)h;
            }
}

// ---------------------------------------------------------------------------
// GEMM2: Out = Hid @ Wdn^T   ([T,I] x [H,I]^T -> [T,H] fp32)
// Dynamic LDS: 2*(BM+BN)*KP*2 = 55296 B
// ---------------------------------------------------------------------------
__global__ __launch_bounds__(256)
void gemm2_down(const bf16* __restrict__ Hid,  // [T, I]
                const bf16* __restrict__ Wdn,  // [H, I]
                float* __restrict__ Out)       // [T, H]
{
    extern __shared__ char smem[];
    bf16* sA = (bf16*)smem;              // [2][BM*KP]
    bf16* sB = sA + 2 * BM * KP;         // [2][BN*KP]

    const int tid  = threadIdx.x;
    const int lane = tid & 31;
    const int wave = tid >> 5;
    const int wm   = wave >> 1;
    const int wn   = wave & 1;
    const int l15  = lane & 15;
    const int hs   = lane >> 4;

    const int m0 = blockIdx.y * BM;
    const int n0 = blockIdx.x * BN;

    v8f acc[2][2];
#pragma unroll
    for (int a = 0; a < 2; ++a)
#pragma unroll
        for (int b = 0; b < 2; ++b) acc[a][b] = (v8f)0.0f;

    const int NT = IDIM / BK;    // 24
    stage2(Hid, Wdn, sA, sB, m0, n0, 0, tid);

    for (int it = 0; it < NT; ++it) {
        const int cur = it & 1;
        bf16* cA = sA + cur * BM * KP;
        bf16* cB = sB + cur * BN * KP;

        if (it + 1 < NT) {
            stage2(Hid, Wdn, sA + (cur ^ 1) * BM * KP, sB + (cur ^ 1) * BN * KP,
                   m0, n0, (it + 1) * BK, tid);
            WAIT_ASYNC(CPT2);
        } else {
            WAIT_ASYNC(0);
        }
        __syncthreads();

#pragma unroll
        for (int ks = 0; ks < BK; ks += 32) {
            v16bf af[2];
#pragma unroll
            for (int mt = 0; mt < 2; ++mt)
                af[mt] = frag_a(cA, wm * 32 + mt * 16 + l15, ks, hs);
#pragma unroll
            for (int nt = 0; nt < 2; ++nt) {
                v16bf bfr = frag_b(cB, wn * 32 + nt * 16 + l15, ks, hs);
#pragma unroll
                for (int mt = 0; mt < 2; ++mt)
                    acc[mt][nt] = WMMA_BF16(af[mt], bfr, acc[mt][nt]);
            }
        }
        __syncthreads();
    }

#pragma unroll
    for (int mt = 0; mt < 2; ++mt)
#pragma unroll
        for (int nt = 0; nt < 2; ++nt)
#pragma unroll
            for (int e = 0; e < 8; ++e) {
                int row = m0 + wm * 32 + mt * 16 + hs * 8 + e;
                int col = n0 + wn * 32 + nt * 16 + l15;
                Out[(long long)row * HDIM + col] = acc[mt][nt][e];
            }
}

// ---------------------------------------------------------------------------
// Host launcher
// ---------------------------------------------------------------------------
extern "C" void kernel_launch(void* const* d_in, const int* in_sizes, int n_in,
                              void* d_out, int out_size, void* d_ws, size_t ws_size,
                              hipStream_t stream)
{
    (void)in_sizes; (void)n_in; (void)out_size; (void)ws_size;

    const float* x    = (const float*)d_in[0];   // [T, H]
    const float* wgu  = (const float*)d_in[1];   // [E, 2I, H]
    const float* wdn  = (const float*)d_in[2];   // [E, H, I]
    const int*   eidx = (const int*)d_in[3];     // scalar
    float*       out  = (float*)d_out;           // [T, H]

    // workspace layout (bf16): x | Wgu[e] | Wdn[e] | hidden   (= 130 MB total)
    char* ws = (char*)d_ws;
    bf16* xb  = (bf16*)ws;
    bf16* gub = xb  + (size_t)TDIM * HDIM;
    bf16* dnb = gub + (size_t)TWOI * HDIM;
    bf16* hid = dnb + (size_t)HDIM * IDIM;

    const long long nx  = (long long)TDIM * HDIM;
    const long long ngu = (long long)TWOI * HDIM;
    const long long ndn = (long long)HDIM * IDIM;

    cvt_f32_to_bf16<<<(unsigned)(nx  / 4 / 256), 256, 0, stream>>>(x,   xb,  nullptr, 0,   nx);
    cvt_f32_to_bf16<<<(unsigned)(ngu / 4 / 256), 256, 0, stream>>>(wgu, gub, eidx,    ngu, ngu);
    cvt_f32_to_bf16<<<(unsigned)(ndn / 4 / 256), 256, 0, stream>>>(wdn, dnb, eidx,    ndn, ndn);

    const size_t lds1 = (size_t)2 * (BM + 2 * BN) * KP * sizeof(bf16);  // 73728
    const size_t lds2 = (size_t)2 * (BM + BN) * KP * sizeof(bf16);      // 55296

    dim3 g1(IDIM / BN, TDIM / BM);   // 24 x 128
    gemm1_swiglu<<<g1, 256, lds1, stream>>>(xb, gub, hid);

    dim3 g2(HDIM / BN, TDIM / BM);   // 32 x 128
    gemm2_down<<<g2, 256, lds2, stream>>>(hid, dnb, out);
}
